// RoIHeads_3762391351654
// MI455X (gfx1250) — matrix operand order repaired and checked by reading, not compiled, not run on hardware
//
#include <hip/hip_runtime.h>
#include <math.h>

#define N_PROP   1000
#define NPAD     1024
#define NCLS     91
#define CM1      90
#define KDET     100
#define SCORE_TH 0.05f
#define NMS_TH   0.5f
#define MIN_SIZE 1.0f
#define BBOX_CLIP 4.135166556742356f   // log(1000/16)

typedef __attribute__((ext_vector_type(16))) _Float16 v16h;
typedef __attribute__((ext_vector_type(8)))  float    v8f;

// ---------------------------------------------------------------------------
// Kernel 1: softmax over 91 classes, one wave32 per proposal row.
// Writes transposed masked-score array score_t[c][i] for classes 1..90
// (class index c in 0..89). Padded rows i in [1000,1024) get -1.
// ---------------------------------------------------------------------------
__global__ void softmax_kernel(const float* __restrict__ logits,
                               float* __restrict__ score_t) {
  const int i = blockIdx.x;
  const int l = threadIdx.x;
  if (i >= N_PROP) {
    for (int c = l; c < CM1; c += 32) score_t[c * NPAD + i] = -1.0f;
    return;
  }
  float x0 = (l      < NCLS) ? logits[i * NCLS + l]      : -3.0e38f;
  float x1 = (l + 32 < NCLS) ? logits[i * NCLS + l + 32] : -3.0e38f;
  float x2 = (l + 64 < NCLS) ? logits[i * NCLS + l + 64] : -3.0e38f;
  float m = fmaxf(x0, fmaxf(x1, x2));
  for (int o = 16; o > 0; o >>= 1) m = fmaxf(m, __shfl_xor(m, o, 32));
  float e0 = (l      < NCLS) ? expf(x0 - m) : 0.0f;
  float e1 = (l + 32 < NCLS) ? expf(x1 - m) : 0.0f;
  float e2 = (l + 64 < NCLS) ? expf(x2 - m) : 0.0f;
  float s = e0 + e1 + e2;
  for (int o = 16; o > 0; o >>= 1) s += __shfl_xor(s, o, 32);
  float inv = 1.0f / s;
  if (l >= 1 && l < NCLS) score_t[(l - 1)  * NPAD + i] = e0 * inv;  // class l
  if (l + 32 < NCLS)      score_t[(l + 31) * NPAD + i] = e1 * inv;  // class l+32
  if (l + 64 < NCLS)      score_t[(l + 63) * NPAD + i] = e2 * inv;  // class l+64
}

// ---------------------------------------------------------------------------
// Kernel 2: per-(class, proposal) box decode + clip + validity mask.
// score_t is updated in place to the masked score (-1 if invalid).
// ---------------------------------------------------------------------------
__global__ void decode_kernel(const float* __restrict__ breg,
                              const float* __restrict__ prop,
                              const int* __restrict__ ihp,
                              const int* __restrict__ iwp,
                              float* __restrict__ score_t,
                              float* __restrict__ boxes) {
  int tid = blockIdx.x * blockDim.x + threadIdx.x;
  if (tid >= CM1 * NPAD) return;
  int c = tid / NPAD, i = tid % NPAD;
  float* bo = &boxes[(c * NPAD + i) * 4];
  if (i >= N_PROP) { bo[0] = bo[1] = bo[2] = bo[3] = 0.0f; return; }

  float W = (float)iwp[0], H = (float)ihp[0];
  float px1 = prop[i * 4 + 0], py1 = prop[i * 4 + 1];
  float px2 = prop[i * 4 + 2], py2 = prop[i * 4 + 3];
  float w = px2 - px1, h = py2 - py1;
  float cx = px1 + 0.5f * w, cy = py1 + 0.5f * h;

  const float* d = &breg[i * (NCLS * 4) + (c + 1) * 4];
  float dx = d[0] * 0.1f, dy = d[1] * 0.1f;
  float dw = fminf(d[2] * 0.2f, BBOX_CLIP);
  float dh = fminf(d[3] * 0.2f, BBOX_CLIP);
  float pcx = dx * w + cx, pcy = dy * h + cy;
  float pw = expf(dw) * w, ph = expf(dh) * h;

  float x1 = fminf(fmaxf(pcx - 0.5f * pw, 0.0f), W);
  float y1 = fminf(fmaxf(pcy - 0.5f * ph, 0.0f), H);
  float x2 = fminf(fmaxf(pcx + 0.5f * pw, 0.0f), W);
  float y2 = fminf(fmaxf(pcy + 0.5f * ph, 0.0f), H);

  float score = score_t[c * NPAD + i];
  bool valid = (score >= SCORE_TH) && ((x2 - x1) >= MIN_SIZE) &&
               ((y2 - y1) >= MIN_SIZE);
  score_t[c * NPAD + i] = valid ? score : -1.0f;
  bo[0] = x1; bo[1] = y1; bo[2] = x2; bo[3] = y2;
}

// ---------------------------------------------------------------------------
// Kernel 3: per-class bitonic argsort (descending by masked score) in LDS.
// ---------------------------------------------------------------------------
__global__ __launch_bounds__(NPAD) void sort_kernel(const float* __restrict__ score_t,
                                                    float* __restrict__ sscore,
                                                    int* __restrict__ sidx) {
  __shared__ float key[NPAD];
  __shared__ int   val[NPAD];
  int c = blockIdx.x, t = threadIdx.x;
  key[t] = -score_t[c * NPAD + t];   // ascending sort of negated = descending
  val[t] = t;
  __syncthreads();
  for (int k = 2; k <= NPAD; k <<= 1) {
    for (int j = k >> 1; j > 0; j >>= 1) {
      int ixj = t ^ j;
      if (ixj > t) {
        float kt = key[t], kx = key[ixj];
        bool up = ((t & k) == 0);
        if (up ? (kt > kx) : (kt < kx)) {
          key[t] = kx; key[ixj] = kt;
          int vt = val[t]; val[t] = val[ixj]; val[ixj] = vt;
        }
      }
      __syncthreads();
    }
  }
  sscore[c * NPAD + t] = -key[t];
  sidx[c * NPAD + t]   = val[t];
}

// ---------------------------------------------------------------------------
// Kernel 4: greedy NMS, one workgroup (1024 threads) per class.
// Sorted boxes are gathered into LDS with CDNA5 async global->LDS b128 copies
// (no VGPR round trip), then the sequential suppression loop runs entirely
// out of LDS, matching the reference fori_loop semantics exactly.
// ---------------------------------------------------------------------------
__global__ __launch_bounds__(NPAD) void nms_kernel(const float* __restrict__ boxes,
                                                   const float* __restrict__ sscore,
                                                   const int* __restrict__ sidx,
                                                   float* __restrict__ keepf) {
  __shared__ float sb[NPAD * 4];
  __shared__ int   skeep[NPAD];
  int c = blockIdx.x, j = threadIdx.x;

  int idx = sidx[c * NPAD + j];
  const float* src = &boxes[(c * NPAD + idx) * 4];
  // Wave-relative LDS byte offset = low 32 bits of the generic shared pointer.
  unsigned lds_off = (unsigned)(size_t)(&sb[j * 4]);
  asm volatile("global_load_async_to_lds_b128 %0, %1, off"
               :: "v"(lds_off), "v"(src) : "memory");
  skeep[j] = (sscore[c * NPAD + j] >= 0.0f) ? 1 : 0;
  asm volatile("s_wait_asynccnt 0x0" ::: "memory");
  __syncthreads();

  float x1j = sb[j * 4 + 0], y1j = sb[j * 4 + 1];
  float x2j = sb[j * 4 + 2], y2j = sb[j * 4 + 3];
  float aj = (x2j - x1j) * (y2j - y1j);

  for (int i = 0; i < NPAD; ++i) {
    if (skeep[i]) {                      // keep[i] as of iteration i (greedy)
      float xi1 = sb[i * 4 + 0], yi1 = sb[i * 4 + 1];
      float xi2 = sb[i * 4 + 2], yi2 = sb[i * 4 + 3];
      if (j > i) {
        float ai = (xi2 - xi1) * (yi2 - yi1);
        float lx = fmaxf(x1j, xi1), ly = fmaxf(y1j, yi1);
        float rx = fminf(x2j, xi2), ry = fminf(y2j, yi2);
        float iw = fmaxf(rx - lx, 0.0f), ihh = fmaxf(ry - ly, 0.0f);
        float inter = iw * ihh;
        float uni = fmaxf(aj + ai - inter, 1e-6f);
        if (inter / uni > NMS_TH) skeep[j] = 0;
      }
    }
    __syncthreads();
  }
  keepf[c * NPAD + j] = (float)skeep[j];
}

// ---------------------------------------------------------------------------
// Kernel 5: compaction + output, one wave32 per class.
// Since scores are sorted descending and kept scores >= 0.05 > -1, top_k of
// (keep ? score : -1) == first min(count,100) kept entries in order.
// Cross-lane exclusive prefix scan of the 32 per-lane keep counts is done
// with two V_WMMA_F32_16X16X32_F16 ops: D = Lstrict(16x32) x t(32x1).
// ---------------------------------------------------------------------------
__global__ __launch_bounds__(32) void output_kernel(const float* __restrict__ boxes,
                                                    const float* __restrict__ sscore,
                                                    const int* __restrict__ sidx,
                                                    const float* __restrict__ keepf,
                                                    float* __restrict__ out) {
  __shared__ float tv[32];    // per-lane keep totals
  __shared__ float offs[32];  // exclusive lane offsets
  int c = blockIdx.x, l = threadIdx.x;

  unsigned mask = 0u;
  for (int m = 0; m < 32; ++m)
    if (keepf[c * NPAD + l * 32 + m] > 0.5f) mask |= (1u << m);
  tv[l] = (float)__popc(mask);
  __syncthreads();

  // A (16x32 f16) layout: lane = M (0..15) x half; VGPR v<4 -> K=2v(+1)+8*half,
  // v>=4 -> K=2v+8(+1)+8*half. B (32x16 f16): column N = lane&15 (only N==0
  // nonzero), lanes 0-15 carry K=0..7,16..23, lanes 16-31 carry K=8..15,24..31 (same order as A).
  v16h a1{}, a2{}, b{};
  int r = l & 15, half = l >> 4;
  for (int v = 0; v < 8; ++v) {
    int K0 = (v < 4 ? 2 * v : 2 * v + 8) + 8 * half;
    a1[2 * v]     = (_Float16)((K0     < r)      ? 1.0f : 0.0f);
    a1[2 * v + 1] = (_Float16)((K0 + 1 < r)      ? 1.0f : 0.0f);
    a2[2 * v]     = (_Float16)((K0     < r + 16) ? 1.0f : 0.0f);
    a2[2 * v + 1] = (_Float16)((K0 + 1 < r + 16) ? 1.0f : 0.0f);
    if (r == 0) {                       // column N==0 holds the t vector
      int kb = (v < 4 ? 2 * v : 2 * v + 8) + 8 * half;
      b[2 * v]     = (_Float16)tv[kb];
      b[2 * v + 1] = (_Float16)tv[kb + 1];
    }
  }
  v8f z{};
  v8f d1 = __builtin_amdgcn_wmma_f32_16x16x32_f16(false, a1, false, b,
                                                  (short)0, z, false, false);
  v8f d2 = __builtin_amdgcn_wmma_f32_16x16x32_f16(false, a2, false, b,
                                                  (short)0, z, false, false);
  // C layout: VGPR q, lanes 0-15 -> (M=q, N=lane); lanes 16-31 -> (M=q+8).
  // Column N==0 lives in lanes 0 and 16.
  if (l == 0)  for (int q = 0; q < 8; ++q) { offs[q]      = d1[q]; offs[16 + q] = d2[q]; }
  if (l == 16) for (int q = 0; q < 8; ++q) { offs[8 + q]  = d1[q]; offs[24 + q] = d2[q]; }
  __syncthreads();

  int rank  = (int)offs[l];
  int count = (int)(offs[31] + tv[31]);

  float* dets   = out;
  float* labels = out + CM1 * KDET * 5;

  for (int m = 0; m < 32; ++m) {
    if (mask & (1u << m)) {
      if (rank < KDET) {
        int j = l * 32 + m;
        int idx = sidx[c * NPAD + j];
        const float* bo = &boxes[(c * NPAD + idx) * 4];
        float* drow = &dets[(c * KDET + rank) * 5];
        drow[0] = bo[0]; drow[1] = bo[1]; drow[2] = bo[2]; drow[3] = bo[3];
        drow[4] = sscore[c * NPAD + j];
        labels[c * KDET + rank] = (float)(c + 1);
      }
      rank++;
    }
  }
  // Deterministically overwrite the empty tail slots every call.
  for (int rr = l; rr < KDET; rr += 32) {
    if (rr >= count) {
      float* drow = &dets[(c * KDET + rr) * 5];
      drow[0] = drow[1] = drow[2] = drow[3] = drow[4] = 0.0f;
      labels[c * KDET + rr] = 0.0f;
    }
  }
}

// ---------------------------------------------------------------------------
extern "C" void kernel_launch(void* const* d_in, const int* in_sizes, int n_in,
                              void* d_out, int out_size, void* d_ws, size_t ws_size,
                              hipStream_t stream) {
  (void)in_sizes; (void)n_in; (void)out_size; (void)ws_size;
  const float* logits = (const float*)d_in[0];   // [1000,91]
  const float* breg   = (const float*)d_in[1];   // [1000,364]
  const float* prop   = (const float*)d_in[2];   // [1000,4]
  const int*   ihp    = (const int*)d_in[3];     // scalar 800
  const int*   iwp    = (const int*)d_in[4];     // scalar 1333
  float* out = (float*)d_out;                    // 45000 dets + 9000 labels

  float* score_t = (float*)d_ws;                 // [90][1024]
  float* boxes   = score_t + CM1 * NPAD;         // [90][1024][4], 16B aligned
  float* sscore  = boxes + CM1 * NPAD * 4;       // [90][1024]
  int*   sidx    = (int*)(sscore + CM1 * NPAD);  // [90][1024]
  float* keepf   = (float*)(sidx + CM1 * NPAD);  // [90][1024]

  softmax_kernel<<<NPAD, 32, 0, stream>>>(logits, score_t);
  decode_kernel<<<(CM1 * NPAD + 255) / 256, 256, 0, stream>>>(
      breg, prop, ihp, iwp, score_t, boxes);
  sort_kernel<<<CM1, NPAD, 0, stream>>>(score_t, sscore, sidx);
  nms_kernel<<<CM1, NPAD, 0, stream>>>(boxes, sscore, sidx, keepf);
  output_kernel<<<CM1, 32, 0, stream>>>(boxes, sscore, sidx, keepf, out);
}
